// Qwen3_5MoeSparseMoeBlockWithMLP_35897336660325
// MI455X (gfx1250) — compile-verified, matrix operand order checked by reading, not run on hardware
//
#include <hip/hip_runtime.h>
#include <hip/hip_bf16.h>

#define DEVINL __device__ __forceinline__

typedef __attribute__((ext_vector_type(16))) __bf16   v16bf;
typedef __attribute__((ext_vector_type(8)))  float    v8f;
typedef __attribute__((ext_vector_type(4)))  unsigned u32x4;
typedef __attribute__((ext_vector_type(4)))  float    f32x4;
typedef int gv4i __attribute__((vector_size(4 * sizeof(int))));  // builtin's pointee type

constexpr int H_DIM  = 2048;  // hidden
constexpr int I_DIM  = 1024;  // expert intermediate
constexpr int E_NUM  = 8;     // experts
constexpr int IS_DIM = 4096;  // shared intermediate
constexpr int T_TOK  = 4096;  // B*S tokens

constexpr int LROW = 40;      // LDS row pitch in halves (32 data + 8 pad -> 20 banks)

// --- CDNA5 async global->LDS DMA (ASYNCcnt), with safe fallback ------------
#if defined(__gfx1250__) && __has_builtin(__builtin_amdgcn_global_load_async_to_lds_b128)
#define ASYNC_LDS 1
#else
#define ASYNC_LDS 0
#endif

DEVINL void async_cp16(const void* g, void* l) {
#if ASYNC_LDS
  __builtin_amdgcn_global_load_async_to_lds_b128(
      (__attribute__((address_space(1))) gv4i*)(g),
      (__attribute__((address_space(3))) gv4i*)(l), 0, 0);
#else
  (void)g; (void)l;
#endif
}

DEVINL void async_wait0() {
#if ASYNC_LDS
#if __has_builtin(__builtin_amdgcn_s_wait_asynccnt)
  __builtin_amdgcn_s_wait_asynccnt(0);
#else
  asm volatile("s_wait_asynccnt 0x0" ::: "memory");
#endif
#endif
}

union FragBF {
  u32x4 q[2];
  v16bf v;
};

DEVINL unsigned cvt2(float a, float b) {
  union { __bf16 h[2]; unsigned u; } r;
  r.h[0] = (__bf16)a;
  r.h[1] = (__bf16)b;
  return r.u;
}

DEVINL float fsilu(float g) { return g / (1.0f + __expf(-g)); }

// ---------------------------------------------------------------------------
// One-time fp32 -> bf16 conversion (memory bound). 8 elements per thread.
// ---------------------------------------------------------------------------
__global__ __launch_bounds__(256)
void cvt_bf16_kernel(const float* __restrict__ in, __bf16* __restrict__ out, long n) {
  const long i = ((long)blockIdx.x * 256 + threadIdx.x) * 8;
  if (i >= n) return;
  f32x4 f0 = *(const f32x4*)(in + i);
  f32x4 f1 = *(const f32x4*)(in + i + 4);
  u32x4 o;
  o.x = cvt2(f0.x, f0.y);
  o.y = cvt2(f0.z, f0.w);
  o.z = cvt2(f1.x, f1.y);
  o.w = cvt2(f1.z, f1.w);
  *(u32x4*)(out + i) = o;
}

// ---------------------------------------------------------------------------
// Router: fp32 logits, softmax, top-2, renorm -> dense combine(T,E); sig(T).
// ---------------------------------------------------------------------------
__global__ __launch_bounds__(256)
void router_kernel(const float* __restrict__ x,
                   const float* __restrict__ w_router,
                   const float* __restrict__ gate_w,
                   float* __restrict__ combine,
                   float* __restrict__ sig) {
  const int t   = blockIdx.x;
  const int tid = threadIdx.x;
  __shared__ float red[9 * 256];
  float acc[9];
#pragma unroll
  for (int a = 0; a < 9; ++a) acc[a] = 0.f;
  const float* xr = x + (size_t)t * H_DIM;
  for (int h = tid; h < H_DIM; h += 256) {
    const float xv = xr[h];
#pragma unroll
    for (int e = 0; e < E_NUM; ++e) acc[e] += xv * w_router[e * H_DIM + h];
    acc[8] += xv * gate_w[h];
  }
#pragma unroll
  for (int a = 0; a < 9; ++a) red[a * 256 + tid] = acc[a];
  __syncthreads();
  for (int s = 128; s > 0; s >>= 1) {
    if (tid < s) {
#pragma unroll
      for (int a = 0; a < 9; ++a) red[a * 256 + tid] += red[a * 256 + tid + s];
    }
    __syncthreads();
  }
  if (tid == 0) {
    float logits[8], mx = -1e30f;
#pragma unroll
    for (int e = 0; e < E_NUM; ++e) { logits[e] = red[e * 256]; mx = fmaxf(mx, logits[e]); }
    float p[8], sum = 0.f;
#pragma unroll
    for (int e = 0; e < E_NUM; ++e) { p[e] = __expf(logits[e] - mx); sum += p[e]; }
#pragma unroll
    for (int e = 0; e < E_NUM; ++e) p[e] /= sum;
    int i0 = 0;
#pragma unroll
    for (int e = 1; e < E_NUM; ++e) if (p[e] > p[i0]) i0 = e;
    int i1 = (i0 == 0) ? 1 : 0;
#pragma unroll
    for (int e = 0; e < E_NUM; ++e) if (e != i0 && p[e] > p[i1]) i1 = e;
    const float norm = p[i0] + p[i1];
#pragma unroll
    for (int e = 0; e < E_NUM; ++e) {
      float c = 0.f;
      if (e == i0) c = p[i0] / norm;
      else if (e == i1) c = p[i1] / norm;
      combine[t * E_NUM + e] = c;
    }
    sig[t] = 1.0f / (1.0f + __expf(-red[8 * 256]));
  }
}

// ---------------------------------------------------------------------------
// Gate+Up SwiGLU GEMM, LDS double-buffered bf16 WMMA (async DMA staging).
//   Out[t, e*Nmat + n] = scale * silu(x@Wg[e]^T) * (x@Wu[e]^T)   (bf16 out)
// Block tile 64(M) x 128(N), K-tile 32; 8 waves, wave tile 16x64.
// ---------------------------------------------------------------------------
__global__ __launch_bounds__(256)
void gateup_wmma_kernel(const __bf16* __restrict__ A,        // (T, K) bf16
                        const __bf16* __restrict__ Wg,       // (Z, Nmat, K) bf16
                        const __bf16* __restrict__ Wu,       // (Z, Nmat, K) bf16
                        __bf16* __restrict__ Out,            // (T, out_stride) bf16
                        const float* __restrict__ combine,   // (T, E) or null
                        int K, int Nmat, int out_stride) {
  __shared__ __bf16 sA[2][64 * LROW];
  __shared__ __bf16 sG[2][128 * LROW];
  __shared__ __bf16 sU[2][128 * LROW];

  const int e    = blockIdx.z;
  const int tid  = threadIdx.x;
  const int lane = tid & 31;
  const int wave = tid >> 5;
  const int hl   = lane >> 4;
  const int q    = lane & 15;
  const int wm   = wave & 3;
  const int wn   = wave >> 2;
  const int rowBase = blockIdx.y * 64;
  const int colBase = blockIdx.x * 128;

  const __bf16* wgmat = Wg + (size_t)e * Nmat * K;
  const __bf16* wumat = Wu + (size_t)e * Nmat * K;

  // cooperative staging maps (one K=32 slice per tile)
  const int aRowL  = tid >> 2;        // 0..63
  const int aChunk = (tid & 3) * 8;   // halves
  const int bRowL  = tid >> 1;        // 0..127
  const int bChunk = (tid & 1) * 16;  // halves

  const __bf16* aG = A     + (size_t)(rowBase + aRowL) * K + aChunk;
  const __bf16* gG = wgmat + (size_t)(colBase + bRowL) * K + bChunk;
  const __bf16* uG = wumat + (size_t)(colBase + bRowL) * K + bChunk;

  __bf16* const dA  = &sA[0][aRowL * LROW + aChunk];
  __bf16* const dG0 = &sG[0][bRowL * LROW + bChunk];
  __bf16* const dU0 = &sU[0][bRowL * LROW + bChunk];
  const int bufA = 64 * LROW;   // halves per A buffer
  const int bufB = 128 * LROW;  // halves per B buffer

#if ASYNC_LDS
  // prologue: DMA tile 0 straight into LDS
  async_cp16(aG, dA);
  async_cp16(gG, dG0);
  async_cp16(gG + 8, dG0 + 8);
  async_cp16(uG, dU0);
  async_cp16(uG + 8, dU0 + 8);
  async_wait0();
  __syncthreads();
#else
  u32x4 rA, rG0, rG1, rU0, rU1;
  rA  = *(const u32x4*)(aG);
  rG0 = *(const u32x4*)(gG);
  rG1 = *(const u32x4*)(gG + 8);
  rU0 = *(const u32x4*)(uG);
  rU1 = *(const u32x4*)(uG + 8);
  *(u32x4*)dA        = rA;
  *(u32x4*)dG0       = rG0;
  *(u32x4*)(dG0 + 8) = rG1;
  *(u32x4*)dU0       = rU0;
  *(u32x4*)(dU0 + 8) = rU1;
  __syncthreads();
#endif

  v8f accG[4], accU[4];
#pragma unroll
  for (int nt = 0; nt < 4; ++nt)
#pragma unroll
    for (int j = 0; j < 8; ++j) { accG[nt][j] = 0.f; accU[nt][j] = 0.f; }

  const int ntiles = K >> 5;
  for (int t = 0; t < ntiles; ++t) {
    const int  cur  = t & 1;
    const int  nxt  = cur ^ 1;
    const bool more = (t + 1) < ntiles;
#if ASYNC_LDS
    if (more) {  // DMA next tile into LDS; overlaps whole compute phase
      const int k0 = (t + 1) << 5;
      async_cp16(aG + k0,     dA + nxt * bufA);
      async_cp16(gG + k0,     dG0 + nxt * bufB);
      async_cp16(gG + k0 + 8, dG0 + nxt * bufB + 8);
      async_cp16(uG + k0,     dU0 + nxt * bufB);
      async_cp16(uG + k0 + 8, dU0 + nxt * bufB + 8);
    }
#else
    if (more) {
      const int k0 = (t + 1) << 5;
      rA  = *(const u32x4*)(aG + k0);
      rG0 = *(const u32x4*)(gG + k0);
      rG1 = *(const u32x4*)(gG + k0 + 8);
      rU0 = *(const u32x4*)(uG + k0);
      rU1 = *(const u32x4*)(uG + k0 + 8);
    }
#endif
    // ---- compute from LDS: load ALL fragments first (distinct regs), then wmma
    FragBF af, bg[4], bu[4];
    {
      const __bf16* pa = &sA[cur][(wm * 16 + q) * LROW + hl * 8];
      af.q[0] = *(const u32x4*)pa;
      af.q[1] = *(const u32x4*)(pa + 16);
    }
#pragma unroll
    for (int nt = 0; nt < 4; ++nt) {
      const __bf16* pg = &sG[cur][(wn * 64 + nt * 16 + q) * LROW + hl * 8];
      bg[nt].q[0] = *(const u32x4*)pg;
      bg[nt].q[1] = *(const u32x4*)(pg + 16);
      const __bf16* pu = &sU[cur][(wn * 64 + nt * 16 + q) * LROW + hl * 8];
      bu[nt].q[0] = *(const u32x4*)pu;
      bu[nt].q[1] = *(const u32x4*)(pu + 16);
    }
#pragma unroll
    for (int nt = 0; nt < 4; ++nt)
      accG[nt] = __builtin_amdgcn_wmma_f32_16x16x32_bf16(false, af.v, false, bg[nt].v,
                                                         (short)0, accG[nt], false, false);
#pragma unroll
    for (int nt = 0; nt < 4; ++nt)
      accU[nt] = __builtin_amdgcn_wmma_f32_16x16x32_bf16(false, af.v, false, bu[nt].v,
                                                         (short)0, accU[nt], false, false);

    if (more) {
#if ASYNC_LDS
      async_wait0();       // own DMA done
      __syncthreads();     // everyone done reading cur & DMA visible
#else
      __syncthreads();
      *(u32x4*)(dA + nxt * bufA)       = rA;
      *(u32x4*)(dG0 + nxt * bufB)      = rG0;
      *(u32x4*)(dG0 + nxt * bufB + 8)  = rG1;
      *(u32x4*)(dU0 + nxt * bufB)      = rU0;
      *(u32x4*)(dU0 + nxt * bufB + 8)  = rU1;
      __syncthreads();
#endif
    }
  }

#pragma unroll
  for (int nt = 0; nt < 4; ++nt) {
    const int col = e * Nmat + colBase + wn * 64 + nt * 16 + q;
#pragma unroll
    for (int v = 0; v < 8; ++v) {
      const int row = rowBase + wm * 16 + v + hl * 8;
      float h = fsilu(accG[nt][v]) * accU[nt][v];
      if (combine) h *= combine[row * E_NUM + e];
      Out[(size_t)row * out_stride + col] = (__bf16)h;
    }
  }
}

// ---------------------------------------------------------------------------
// Down-projection GEMM, LDS double-buffered bf16 WMMA (async DMA staging):
//   C[t,n] (+)= rowscale[t] * sum_k A[t,k] * B[k>>il][n][k&mask]
// Block tile 128(M) x 128(N), K-tile 32; wave tile 32x64.
// ---------------------------------------------------------------------------
__global__ __launch_bounds__(256)
void down_wmma_kernel(const __bf16* __restrict__ A,         // (T, K) bf16
                      const __bf16* __restrict__ B,         // bf16 weights
                      float* __restrict__ C,                // (T, N) fp32
                      const float* __restrict__ rowscale,   // (T) or null
                      int accumulate,
                      int K, int inner_log2, size_t e_stride, int n_stride, int N) {
  __shared__ __bf16 sA[2][128 * LROW];
  __shared__ __bf16 sB[2][128 * LROW];

  const int tid  = threadIdx.x;
  const int lane = tid & 31;
  const int wave = tid >> 5;
  const int hl   = lane >> 4;
  const int q    = lane & 15;
  const int wm   = wave & 3;
  const int wn   = wave >> 2;
  const int rowBase = blockIdx.y * 128;
  const int colBase = blockIdx.x * 128;
  const int innerMask = (1 << inner_log2) - 1;

  const int rowL  = tid >> 1;         // 0..127
  const int chunk = (tid & 1) * 16;   // halves

  const __bf16* aG = A + (size_t)(rowBase + rowL) * K + chunk;
  const __bf16* bG = B + (size_t)(colBase + rowL) * n_stride + chunk;

  __bf16* const dA = &sA[0][rowL * LROW + chunk];
  __bf16* const dB = &sB[0][rowL * LROW + chunk];
  const int bufH = 128 * LROW;  // halves per buffer

#if ASYNC_LDS
  async_cp16(aG,     dA);
  async_cp16(aG + 8, dA + 8);
  async_cp16(bG,     dB);
  async_cp16(bG + 8, dB + 8);
  async_wait0();
  __syncthreads();
#else
  u32x4 rA0, rA1, rB0, rB1;
  rA0 = *(const u32x4*)(aG);
  rA1 = *(const u32x4*)(aG + 8);
  rB0 = *(const u32x4*)(bG);
  rB1 = *(const u32x4*)(bG + 8);
  *(u32x4*)dA       = rA0;
  *(u32x4*)(dA + 8) = rA1;
  *(u32x4*)dB       = rB0;
  *(u32x4*)(dB + 8) = rB1;
  __syncthreads();
#endif

  v8f acc[2][4];
#pragma unroll
  for (int mi = 0; mi < 2; ++mi)
#pragma unroll
    for (int nt = 0; nt < 4; ++nt)
#pragma unroll
      for (int j = 0; j < 8; ++j) acc[mi][nt][j] = 0.f;

  const int ntiles = K >> 5;
  for (int t = 0; t < ntiles; ++t) {
    const int  cur  = t & 1;
    const int  nxt  = cur ^ 1;
    const bool more = (t + 1) < ntiles;
    // K-tiles of 32 never straddle an expert boundary (I % 32 == 0)
#if ASYNC_LDS
    if (more) {
      const int k0 = (t + 1) << 5;
      const size_t boff = (size_t)(k0 >> inner_log2) * e_stride + (k0 & innerMask);
      async_cp16(aG + k0,       dA + nxt * bufH);
      async_cp16(aG + k0 + 8,   dA + nxt * bufH + 8);
      async_cp16(bG + boff,     dB + nxt * bufH);
      async_cp16(bG + boff + 8, dB + nxt * bufH + 8);
    }
#else
    if (more) {
      const int k0 = (t + 1) << 5;
      const size_t boff = (size_t)(k0 >> inner_log2) * e_stride + (k0 & innerMask);
      rA0 = *(const u32x4*)(aG + k0);
      rA1 = *(const u32x4*)(aG + k0 + 8);
      rB0 = *(const u32x4*)(bG + boff);
      rB1 = *(const u32x4*)(bG + boff + 8);
    }
#endif
    // ---- load all fragments first (distinct regs), then the 8 wmma
    FragBF a0, a1, bf[4];
    {
      const __bf16* p0 = &sA[cur][(wm * 32 + q) * LROW + hl * 8];
      const __bf16* p1 = &sA[cur][(wm * 32 + 16 + q) * LROW + hl * 8];
      a0.q[0] = *(const u32x4*)p0;
      a0.q[1] = *(const u32x4*)(p0 + 16);
      a1.q[0] = *(const u32x4*)p1;
      a1.q[1] = *(const u32x4*)(p1 + 16);
    }
#pragma unroll
    for (int nt = 0; nt < 4; ++nt) {
      const __bf16* pb = &sB[cur][(wn * 64 + nt * 16 + q) * LROW + hl * 8];
      bf[nt].q[0] = *(const u32x4*)pb;
      bf[nt].q[1] = *(const u32x4*)(pb + 16);
    }
#pragma unroll
    for (int nt = 0; nt < 4; ++nt) {
      acc[0][nt] = __builtin_amdgcn_wmma_f32_16x16x32_bf16(false, a0.v, false, bf[nt].v,
                                                           (short)0, acc[0][nt], false, false);
      acc[1][nt] = __builtin_amdgcn_wmma_f32_16x16x32_bf16(false, a1.v, false, bf[nt].v,
                                                           (short)0, acc[1][nt], false, false);
    }

    if (more) {
#if ASYNC_LDS
      async_wait0();
      __syncthreads();
#else
      __syncthreads();
      *(u32x4*)(dA + nxt * bufH)     = rA0;
      *(u32x4*)(dA + nxt * bufH + 8) = rA1;
      *(u32x4*)(dB + nxt * bufH)     = rB0;
      *(u32x4*)(dB + nxt * bufH + 8) = rB1;
      __syncthreads();
#endif
    }
  }

#pragma unroll
  for (int mi = 0; mi < 2; ++mi) {
#pragma unroll
    for (int nt = 0; nt < 4; ++nt) {
      const int col = colBase + wn * 64 + nt * 16 + q;
#pragma unroll
      for (int v = 0; v < 8; ++v) {
        const int row = rowBase + wm * 32 + mi * 16 + v + hl * 8;
        float val = acc[mi][nt][v];
        if (rowscale) val *= rowscale[row];
        const size_t idx = (size_t)row * N + col;
        if (accumulate) C[idx] += val;
        else            C[idx]  = val;
      }
    }
  }
}

// ---------------------------------------------------------------------------
// Host launch
// ---------------------------------------------------------------------------
static inline void launch_cvt(const float* in, __bf16* out, long n, hipStream_t s) {
  const long blocks = n / (256L * 8L);
  cvt_bf16_kernel<<<dim3((unsigned)blocks), 256, 0, s>>>(in, out, n);
}

extern "C" void kernel_launch(void* const* d_in, const int* in_sizes, int n_in,
                              void* d_out, int out_size, void* d_ws, size_t ws_size,
                              hipStream_t stream) {
  (void)in_sizes; (void)n_in; (void)out_size; (void)ws_size;
  const float* x        = (const float*)d_in[0];  // (T, H)
  const float* w_router = (const float*)d_in[1];  // (E, H)
  const float* wg       = (const float*)d_in[2];  // (E, I, H)
  const float* wu       = (const float*)d_in[3];  // (E, I, H)
  const float* wd       = (const float*)d_in[4];  // (E, H, I)
  const float* sw_gate  = (const float*)d_in[5];  // (IS, H)
  const float* sw_up    = (const float*)d_in[6];  // (IS, H)
  const float* sw_down  = (const float*)d_in[7];  // (H, IS)
  const float* gate_w   = (const float*)d_in[8];  // (1, H)
  float* out = (float*)d_out;                     // (T, H)

  // Workspace layout (16B-aligned chunks)
  char* w = (char*)d_ws;
  float* combine = (float*)w;  w += (size_t)T_TOK * E_NUM * sizeof(float);
  float* sig     = (float*)w;  w += (size_t)T_TOK * sizeof(float);
  __bf16* xb   = (__bf16*)w;   w += (size_t)T_TOK * H_DIM * 2;          // 16MB
  __bf16* sgb  = (__bf16*)w;   w += (size_t)IS_DIM * H_DIM * 2;         // 16MB
  __bf16* sub  = (__bf16*)w;   w += (size_t)IS_DIM * H_DIM * 2;         // 16MB
  __bf16* sdb  = (__bf16*)w;   w += (size_t)H_DIM * IS_DIM * 2;         // 16MB
  __bf16* wgb  = (__bf16*)w;   w += (size_t)E_NUM * I_DIM * H_DIM * 2;  // 32MB
  __bf16* wub  = (__bf16*)w;   w += (size_t)E_NUM * I_DIM * H_DIM * 2;  // 32MB
  __bf16* wdb  = (__bf16*)w;   w += (size_t)E_NUM * H_DIM * I_DIM * 2;  // 32MB
  __bf16* Hs   = (__bf16*)w;   w += (size_t)T_TOK * IS_DIM * 2;         // 32MB
  __bf16* Hexp = (__bf16*)w;   w += (size_t)T_TOK * E_NUM * I_DIM * 2;  // 64MB

  // 0) one-time fp32 -> bf16 conversions
  launch_cvt(x,       xb,  (long)T_TOK * H_DIM, stream);
  launch_cvt(sw_gate, sgb, (long)IS_DIM * H_DIM, stream);
  launch_cvt(sw_up,   sub, (long)IS_DIM * H_DIM, stream);
  launch_cvt(sw_down, sdb, (long)H_DIM * IS_DIM, stream);
  launch_cvt(wg,      wgb, (long)E_NUM * I_DIM * H_DIM, stream);
  launch_cvt(wu,      wub, (long)E_NUM * I_DIM * H_DIM, stream);
  launch_cvt(wd,      wdb, (long)E_NUM * H_DIM * I_DIM, stream);

  // 1) router -> combine(T,E), sig(T)
  router_kernel<<<T_TOK, 256, 0, stream>>>(x, w_router, gate_w, combine, sig);

  // 2) shared-expert gate/up SwiGLU: Hs = silu(x sw_gate^T) * (x sw_up^T)
  gateup_wmma_kernel<<<dim3(IS_DIM / 128, T_TOK / 64, 1), 256, 0, stream>>>(
      xb, sgb, sub, Hs, nullptr, H_DIM, IS_DIM, IS_DIM);

  // 3) expert gate/up SwiGLU scaled by combine: Hexp (T, E*I)
  gateup_wmma_kernel<<<dim3(I_DIM / 128, T_TOK / 64, E_NUM), 256, 0, stream>>>(
      xb, wgb, wub, Hexp, combine, H_DIM, I_DIM, E_NUM * I_DIM);

  // 4) shared down: out = sig[t] * (Hs @ sw_down^T)
  down_wmma_kernel<<<dim3(H_DIM / 128, T_TOK / 128, 1), 256, 0, stream>>>(
      Hs, sdb, out, sig, /*accumulate=*/0,
      IS_DIM, /*inner_log2=*/12, /*e_stride=*/0, /*n_stride=*/IS_DIM, H_DIM);

  // 5) expert down: out += Hexp @ wd'
  down_wmma_kernel<<<dim3(H_DIM / 128, T_TOK / 128, 1), 256, 0, stream>>>(
      Hexp, wdb, out, nullptr, /*accumulate=*/1,
      E_NUM * I_DIM, /*inner_log2=*/10, /*e_stride=*/(size_t)H_DIM * I_DIM,
      /*n_stride=*/I_DIM, H_DIM);
}
